// MultiHeadAttention_8126078124686
// MI455X (gfx1250) — matrix operand order repaired and checked by reading, not compiled, not run on hardware
//
#include <hip/hip_runtime.h>
#include <hip/hip_bf16.h>

// ---------------------------------------------------------------------------
// Multi-head self-attention for MI455X (gfx1250, wave32, WMMA).
//   B=4, S=2048, E=512, H=8, Dh=64
// d_out = [ out (B,S,E) f32 | attention (H,B,S,S) f32 ]
//
// Pipeline:
//   cvt (f32->f16, weights pre-transposed) -> QKV WMMA GEMMs ->
//   fused {scores -> mask -> softmax -> NT prob writeback -> P@V} per
//   16-query strip held in LDS (128 KB) -> output-projection WMMA GEMM.
// ---------------------------------------------------------------------------

#define BATCH   4
#define SEQ     2048
#define EMB     512
#define NHEAD   8
#define HDIM    64
#define BS      (BATCH * SEQ)        // 8192 rows
#define NEGBIG  (-100000000.0f)
#define NORM    0.125f               // 1/sqrt(64)

typedef __attribute__((ext_vector_type(16))) _Float16 v16h;
typedef __attribute__((ext_vector_type(8)))  _Float16 v8h;
typedef __attribute__((ext_vector_type(8)))  float    v8f;
typedef __attribute__((ext_vector_type(4)))  float    v4f;

union HF16 { v16h v; v8h h8[2]; _Float16 e[16]; };

// ---- WMMA wrapper: D = A(16x32 f16) * B(32x16 f16) + C(f32) ----
__device__ __forceinline__ v8f wmma16(v16h a, v16h b, v8f c) {
    return __builtin_amdgcn_wmma_f32_16x16x32_f16(
        /*neg_a=*/false, a, /*neg_b=*/false, b,
        /*c_mod=*/(short)0, c, /*reuse_a=*/false, /*reuse_b=*/false);
}

// A fragment (16x32 f16) from row-major [M][ld] f16 matrix.
// lane L -> row L&15; a[0..7]=K k0+(L>>4)*8+j ; a[8..15]=K k0+16+(L>>4)*8+j
__device__ __forceinline__ v16h load_a16(const _Float16* base, int ld,
                                         int row0, int k0, int lane) {
    const int half = lane >> 4, r = lane & 15;
    const _Float16* p = base + (size_t)(row0 + r) * ld + k0 + half * 8;
    HF16 f;
    f.h8[0] = *(const v8h*)(p);
    f.h8[1] = *(const v8h*)(p + 16);
    return f.v;
}

// B fragment (32x16 f16) from "transposed" operand stored row-major [N][ld].
// lane L -> col n0+(L&15); b[0..7]=K k0+(L>>4)*8+j ; b[8..15]=K k0+16+(L>>4)*8+j
__device__ __forceinline__ v16h load_b16(const _Float16* baseT, int ld,
                                         int n0, int k0, int lane) {
    const int kg = lane >> 4, c = lane & 15;
    const _Float16* p = baseT + (size_t)(n0 + c) * ld + k0 + kg * 8;
    HF16 f;
    f.h8[0] = *(const v8h*)(p);
    f.h8[1] = *(const v8h*)(p + 16);
    return f.v;
}

// A fragment from a row-major f32 matrix (e.g. LDS prob strip), scaled by
// per-row `scale`, converted f16 on the fly.
__device__ __forceinline__ v16h load_a32cvt_scale(const float* base, int ld,
                                                  int row0, int k0, int lane,
                                                  float scale) {
    const int half = lane >> 4, r = lane & 15;
    const float* p = base + (size_t)(row0 + r) * ld + k0 + half * 8;
    HF16 f;
#pragma unroll
    for (int j = 0; j < 8; ++j) {
        f.e[j]     = (_Float16)(p[j] * scale);
        f.e[8 + j] = (_Float16)(p[16 + j] * scale);
    }
    return f.v;
}

// ---------------------------------------------------------------------------
// K0a: f32 -> f16 elementwise (activations)
// ---------------------------------------------------------------------------
__global__ void cvt_x_kernel(const float* __restrict__ src,
                             _Float16* __restrict__ dst, int n) {
    int i = blockIdx.x * blockDim.x + threadIdx.x;
    if (i < n) dst[i] = (_Float16)src[i];
}

// K0b: QKV weight [H][E][Dh] f32 -> Wt [N=h*64+k][E] f16 (transposed for B-frags)
__global__ void cvt_wqkv_kernel(const float* __restrict__ W,
                                _Float16* __restrict__ Wt) {
    int i = blockIdx.x * blockDim.x + threadIdx.x;   // over 512*512
    int n = i >> 9, d = i & 511;
    int h = n >> 6, k = n & 63;
    Wt[i] = (_Float16)W[((size_t)h * EMB + d) * HDIM + k];
}

// K0c: out weight [H][Dh][E] f32 -> Wot [E][h*64+v] f16 (transposed for B-frags)
__global__ void cvt_wout_kernel(const float* __restrict__ W,
                                _Float16* __restrict__ Wt) {
    int i = blockIdx.x * blockDim.x + threadIdx.x;   // over 512*512
    int e = i >> 9, hv = i & 511;
    int h = hv >> 6, v = hv & 63;
    Wt[i] = (_Float16)W[((size_t)h * HDIM + v) * EMB + e];
}

// ---------------------------------------------------------------------------
// K1: C[8192x512] = A[8192x512] * B (Bt given as [512][512] transposed), f16.
// mode 0: C row-major [M][512]   (Q, K projections)
// mode 1: C transposed [N][8192] (V projection -> Vt for PV GEMM B-frags)
// grid(512,4), block 256 (8 waves, 1 tile each)
// ---------------------------------------------------------------------------
__global__ void gemm_qkv_kernel(const _Float16* __restrict__ A,
                                const _Float16* __restrict__ Bt,
                                _Float16* __restrict__ C, int mode) {
    const int lane = threadIdx.x & 31;
    const int wave = threadIdx.x >> 5;
    const int m0 = blockIdx.x * 16;
    const int n0 = (blockIdx.y * 8 + wave) * 16;
    v8f c = {};
    for (int k0 = 0; k0 < EMB; k0 += 32) {
        if (k0 + 32 < EMB)
            __builtin_prefetch(A + (size_t)m0 * EMB + k0 + 32, 0, 3);
        v16h a = load_a16(A, EMB, m0, k0, lane);
        v16h b = load_b16(Bt, EMB, n0, k0, lane);
        c = wmma16(a, b, c);
    }
    const int r = lane >> 4, col = lane & 15;
    if (mode == 0) {
        _Float16* p = C + (size_t)(m0 + r * 8) * EMB + n0 + col;
#pragma unroll
        for (int i = 0; i < 8; ++i) p[(size_t)i * EMB] = (_Float16)c[i];
    } else {
        _Float16* p = C + (size_t)(n0 + col) * BS + m0 + r * 8;
#pragma unroll
        for (int i = 0; i < 8; ++i) p[i] = (_Float16)c[i];
    }
}

// ---------------------------------------------------------------------------
// K2 (fused): for one (h,b, 16-query strip):
//   scores (WMMA) -> LDS, mask, softmax (shfl width-16 reductions),
//   non-temporal writeback of final probabilities, P@V (A from LDS).
// grid(128, 32), block 256 (8 waves), dynamic LDS = 132 KB.
// ---------------------------------------------------------------------------
__global__ void attn_fused_kernel(const _Float16* __restrict__ Qh,
                                  const _Float16* __restrict__ Kh,
                                  const _Float16* __restrict__ Vt,
                                  const int* __restrict__ mask,
                                  float* __restrict__ attn,
                                  _Float16* __restrict__ Hh) {
    extern __shared__ char smem[];
    float* Pr      = (float*)smem;            // [16][2048] score/exp strip
    float* invArr  = Pr + 16 * SEQ;           // [16] per-row 1/sum
    float* partial = invArr + 16;             // [4][256] cross-wave C scratch

    const int t    = threadIdx.x;
    const int lane = t & 31;
    const int wave = t >> 5;                  // 0..7
    const int qt   = blockIdx.x;              // 0..127
    const int hb   = blockIdx.y;              // h*B + b, 0..31
    const int h    = hb >> 2, b = hb & 3;

    const _Float16* Qb = Qh + (size_t)b * SEQ * EMB;
    const _Float16* Kb = Kh + (size_t)b * SEQ * EMB;

    // ---- Phase 1: masked scaled scores into LDS ----
    {
        // Q fragments are invariant across key tiles: hoist.
        v16h a0 = load_a16(Qb, EMB, qt * 16, h * HDIM + 0, lane);
        v16h a1 = load_a16(Qb, EMB, qt * 16, h * HDIM + 32, lane);
        const int col = lane & 15, r = lane >> 4;
        for (int kt = wave; kt < SEQ / 16; kt += 8) {
            v16h b0 = load_b16(Kb, EMB, kt * 16, h * HDIM + 0, lane);
            v16h b1 = load_b16(Kb, EMB, kt * 16, h * HDIM + 32, lane);
            v8f c = {};
            c = wmma16(a0, b0, c);
            c = wmma16(a1, b1, c);
            const int ki = kt * 16 + col;
            const int km = mask[b * SEQ + ki];
#pragma unroll
            for (int i = 0; i < 8; ++i) {
                float v = (km > 0) ? NEGBIG : c[i] * NORM;
                Pr[(r * 8 + i) * SEQ + ki] = v;
            }
        }
    }
    __syncthreads();

    // ---- Phase 2: softmax stats; LDS keeps exp (unnormalized) ----
    {
        const int row = t >> 4, sub = t & 15;  // 16 threads per row (half-wave)
        float* rowp = Pr + row * SEQ;
        float m = -3.0e38f;
        for (int j = sub; j < SEQ; j += 16) m = fmaxf(m, rowp[j]);
#pragma unroll
        for (int off = 8; off >= 1; off >>= 1)
            m = fmaxf(m, __shfl_xor(m, off, 16));
        float s = 0.0f;
        for (int j = sub; j < SEQ; j += 16) {
            float e = __expf(rowp[j] - m);
            rowp[j] = e;
            s += e;
        }
#pragma unroll
        for (int off = 8; off >= 1; off >>= 1)
            s += __shfl_xor(s, off, 16);
        if (sub == 0) invArr[row] = 1.0f / s;
    }
    __syncthreads();

    // ---- Phase 2b: coalesced non-temporal writeback of probabilities ----
    {
        float* gbase = attn + ((size_t)hb * SEQ + qt * 16) * SEQ;
        for (int idx = t; idx < 16 * (SEQ / 4); idx += 256) {
            const int row4 = idx >> 9;         // SEQ/4 = 512 float4 per row
            const int c4   = idx & 511;
            v4f v = *(const v4f*)(Pr + row4 * SEQ + c4 * 4);
            v *= invArr[row4];
            __builtin_nontemporal_store(v, (v4f*)(gbase + (size_t)row4 * SEQ + c4 * 4));
        }
    }

    // ---- Phase 3: heads strip = P @ V (A from LDS, scaled by 1/sum) ----
    {
        const int ntile = wave & 3;            // Dh tile 0..3
        const int khalf = wave >> 2;           // split K across wave pairs
        const float invR = invArr[lane & 15];  // A-frag row = lane&15
        v8f c = {};
        for (int k0 = khalf * (SEQ / 2); k0 < (khalf + 1) * (SEQ / 2); k0 += 32) {
            v16h a  = load_a32cvt_scale(Pr, SEQ, 0, k0, lane, invR);
            v16h bb = load_b16(Vt, BS, h * HDIM + ntile * 16, b * SEQ + k0, lane);
            c = wmma16(a, bb, c);
        }
        if (wave >= 4) {
#pragma unroll
            for (int i = 0; i < 8; ++i) partial[ntile * 256 + lane * 8 + i] = c[i];
        }
        __syncthreads();
        if (wave < 4) {
            const int r = lane >> 4, col = lane & 15;
            _Float16* p = Hh + ((size_t)b * SEQ + qt * 16 + r * 8) * EMB
                             + h * HDIM + ntile * 16 + col;
#pragma unroll
            for (int i = 0; i < 8; ++i)
                p[(size_t)i * EMB] = (_Float16)(c[i] + partial[ntile * 256 + lane * 8 + i]);
        }
    }
}

// ---------------------------------------------------------------------------
// K3: out[8192x512] f32 = heads[8192x512] @ Wout (Wot transposed [512][512]).
// grid(512,4), block 256.
// ---------------------------------------------------------------------------
__global__ void gemm_out_kernel(const _Float16* __restrict__ A,
                                const _Float16* __restrict__ Bt,
                                float* __restrict__ C) {
    const int lane = threadIdx.x & 31;
    const int wave = threadIdx.x >> 5;
    const int m0 = blockIdx.x * 16;
    const int n0 = (blockIdx.y * 8 + wave) * 16;
    v8f c = {};
    for (int k0 = 0; k0 < EMB; k0 += 32) {
        if (k0 + 32 < EMB)
            __builtin_prefetch(A + (size_t)m0 * EMB + k0 + 32, 0, 3);
        v16h a = load_a16(A, EMB, m0, k0, lane);
        v16h b = load_b16(Bt, EMB, n0, k0, lane);
        c = wmma16(a, b, c);
    }
    const int r = lane >> 4, col = lane & 15;
    float* p = C + (size_t)(m0 + r * 8) * EMB + n0 + col;
#pragma unroll
    for (int i = 0; i < 8; ++i) p[(size_t)i * EMB] = c[i];
}

// ---------------------------------------------------------------------------
extern "C" void kernel_launch(void* const* d_in, const int* in_sizes, int n_in,
                              void* d_out, int out_size, void* d_ws, size_t ws_size,
                              hipStream_t stream) {
    const float* q    = (const float*)d_in[0];
    const int*   mask = (const int*)d_in[1];
    const float* wq   = (const float*)d_in[2];
    const float* wk   = (const float*)d_in[3];
    const float* wv   = (const float*)d_in[4];
    const float* wo   = (const float*)d_in[5];

    float* out  = (float*)d_out;                         // [B,S,E]
    float* attn = out + (size_t)BATCH * SEQ * EMB;       // [H,B,S,S]

    // Workspace carve-up (f16 buffers), ~42 MB total.
    char* ws = (char*)d_ws;
    size_t off = 0;
    _Float16* Xh  = (_Float16*)(ws + off); off += (size_t)BS * EMB * 2;
    _Float16* Wqt = (_Float16*)(ws + off); off += (size_t)EMB * EMB * 2;
    _Float16* Wkt = (_Float16*)(ws + off); off += (size_t)EMB * EMB * 2;
    _Float16* Wvt = (_Float16*)(ws + off); off += (size_t)EMB * EMB * 2;
    _Float16* Wot = (_Float16*)(ws + off); off += (size_t)EMB * EMB * 2;
    _Float16* Qh  = (_Float16*)(ws + off); off += (size_t)BS * EMB * 2;   // [m][512]
    _Float16* Kh  = (_Float16*)(ws + off); off += (size_t)BS * EMB * 2;   // [m][512]
    _Float16* Vt  = (_Float16*)(ws + off); off += (size_t)EMB * BS * 2;   // [n][8192]
    _Float16* Hh  = (_Float16*)(ws + off); off += (size_t)BS * EMB * 2;   // [m][512]

    // --- conversions ---
    {
        int n = BS * EMB;
        cvt_x_kernel<<<(n + 255) / 256, 256, 0, stream>>>(q, Xh, n);
        cvt_wqkv_kernel<<<(EMB * EMB) / 256, 256, 0, stream>>>(wq, Wqt);
        cvt_wqkv_kernel<<<(EMB * EMB) / 256, 256, 0, stream>>>(wk, Wkt);
        cvt_wqkv_kernel<<<(EMB * EMB) / 256, 256, 0, stream>>>(wv, Wvt);
        cvt_wout_kernel<<<(EMB * EMB) / 256, 256, 0, stream>>>(wo, Wot);
    }

    // --- QKV projections (WMMA GEMMs) ---
    {
        dim3 grid(BS / 16, 4);
        gemm_qkv_kernel<<<grid, 256, 0, stream>>>(Xh, Wqt, Qh, 0);
        gemm_qkv_kernel<<<grid, 256, 0, stream>>>(Xh, Wkt, Kh, 0);
        gemm_qkv_kernel<<<grid, 256, 0, stream>>>(Xh, Wvt, Vt, 1);
    }

    // --- fused scores/softmax/prob-writeback/PV ---
    {
        dim3 grid(SEQ / 16, NHEAD * BATCH);
        size_t smem = (size_t)(16 * SEQ + 16 + 4 * 256) * sizeof(float); // 132 KB
        attn_fused_kernel<<<grid, 256, smem, stream>>>(Qh, Kh, Vt, mask, attn, Hh);
    }

    // --- output projection ---
    {
        dim3 grid(BS / 16, 4);
        gemm_out_kernel<<<grid, 256, 0, stream>>>(Hh, Wot, out);
    }
}
